// Encoding_76794015252631
// MI455X (gfx1250) — matrix operand order repaired and checked by reading, not compiled, not run on hardware
//
#include <hip/hip_runtime.h>
#include <hip/hip_bf16.h>
#include <math.h>

// ---------------------------------------------------------------------------
// Problem constants: B=64, T=2048, D=300.  M = B*T = 131072 rows.
// GEMM: z = tanh(X @ Wz^T + bz), o = tanh(X @ Wo^T + bo)
// Scan: c_t = g*c + (1-g)*z ; h = o*c   (affine-composable -> chunked scan)
// ---------------------------------------------------------------------------
#define BB   64
#define TT   2048
#define DD   300
#define MM   (BB * TT)          // 131072
#define KPAD 320                // K padded to 10 x 32 (WMMA K-step)
#define NT   19                 // ceil(300/16) output column tiles
#define KT   10                 // K steps of 32
#define MROWS 64                // A-stripe rows per workgroup (4 waves x 16)
#define NCH  16                 // scan chunks
#define CHL  (TT / NCH)         // 128 timesteps per chunk

typedef __attribute__((ext_vector_type(16))) __bf16 v16bf;
typedef __attribute__((ext_vector_type(8)))  float  v8f;

union Frag { v16bf v; unsigned u[8]; };

// ---------------------------------------------------------------------------
// Kernel 0: fused dual GEMM + bias + tanh via v_wmma_f32_16x16x32_bf16.
// One block (128 thr = 4 wave32) owns a 64-row stripe of X, staged once into
// LDS as bf16, then sweeps all 19 N-tiles so X is read from HBM exactly once.
// Weight tiles (both Wz and Wo) are staged to LDS per N-tile and shared by
// all 4 waves.  A-fragments for all 10 K-steps are hoisted into registers.
// B-fragments are software-pipelined two K-groups deep so ds_load latency
// overlaps WMMA issue; four accumulator chains (z/o x even/odd K) avoid
// WMMA->WMMA RAW bubbles (ISA 7.12.1).
// ---------------------------------------------------------------------------
__global__ __launch_bounds__(128)
void gated_gemm_kernel(const float* __restrict__ X,
                       const float* __restrict__ Wz,
                       const float* __restrict__ Wo,
                       const float* __restrict__ bz,
                       const float* __restrict__ bo,
                       float* __restrict__ zws,
                       float* __restrict__ ows) {
    __shared__ __align__(16) __bf16 ldsA[MROWS * KPAD];     // 40 KB
    __shared__ __align__(16) __bf16 ldsW[2][16 * KPAD];     // 2 x 10 KB

    const int tid  = threadIdx.x;
    const int lane = tid & 31;
    const int wv   = tid >> 5;          // wave id 0..3 -> M sub-tile
    const int lm   = lane & 15;         // row/col within fragment
    const int lhi  = lane >> 4;         // half-wave select
    const int m0   = blockIdx.x * MROWS;

    // ---- stage A stripe: 64 x 300 fp32 -> bf16 in LDS (zero-pad K) --------
    for (int idx = tid; idx < MROWS * KPAD; idx += 128) {
        int r = idx / KPAD, c = idx - r * KPAD;
        float v = (c < DD) ? X[(size_t)(m0 + r) * DD + c] : 0.0f;
        ldsA[idx] = (__bf16)v;
    }
    __syncthreads();

    // ---- hoist all 10 A-fragments into registers (contiguous k = 16h + s order) ----
    // lane(0..15): rows M=0..15, K = 0..15; lane(16..31): K+16
    Frag afrag[KT];
    {
        const __bf16* arow = &ldsA[(wv * 16 + lm) * KPAD];
        const int koff = lhi * 16;
#pragma unroll
        for (int kt = 0; kt < KT; ++kt) {
#pragma unroll
            for (int p = 0; p < 8; ++p) {
                int k = kt * 32 + koff + 2 * p;
                afrag[kt].u[p] = *(const unsigned*)(arow + k);
            }
        }
    }

    // B layout: lane n=lm holds 16 contiguous K for column n;
    // lanes 16..31 hold K+16 -> two 16B contiguous LDS reads per frag.
    const int koff2 = lhi * 16;

    // ---- sweep output column tiles ---------------------------------------
    for (int nt = 0; nt < NT; ++nt) {
        const int n0 = nt * 16;

        // stage 16 x K weight tiles for BOTH matrices (bf16, zero-padded)
        for (int idx = tid; idx < 16 * KPAD; idx += 128) {
            int r = idx / KPAD, c = idx - r * KPAD;
            int e = n0 + r;
            float vz = 0.0f, vo = 0.0f;
            if (e < DD && c < DD) {
                vz = Wz[e * DD + c];
                vo = Wo[e * DD + c];
            }
            ldsW[0][idx] = (__bf16)vz;
            ldsW[1][idx] = (__bf16)vo;
        }

        // speculative prefetch of the NEXT weight tile into the WGP cache
        // (global_prefetch_b8) while this tile is being consumed below.
        if (nt + 1 < NT) {
            int e = n0 + 16 + (tid >> 3);        // 16 rows, 8 threads/row
            int c = (tid & 7) * 40;              // one prefetch per 160 B
            if (e < DD && c < DD) {
                __builtin_prefetch(&Wz[e * DD + c], 0, 1);
                __builtin_prefetch(&Wo[e * DD + c], 0, 1);
            }
        }
        __syncthreads();

        v8f accz0, accz1, acco0, acco1;
#pragma unroll
        for (int r = 0; r < 8; ++r) {
            accz0[r] = 0.0f; accz1[r] = 0.0f;
            acco0[r] = 0.0f; acco1[r] = 0.0f;
        }

        const __bf16* rz = &ldsW[0][lm * KPAD + koff2];
        const __bf16* ro = &ldsW[1][lm * KPAD + koff2];

        // ---- software-pipelined K loop (2 K-groups in flight) -------------
        Frag cz0, co0, cz1, co1;     // current group (kt, kt+1)
#pragma unroll
        for (int p = 0; p < 8; ++p) {
            cz0.u[p] = *(const unsigned*)(rz + 2 * p);
            co0.u[p] = *(const unsigned*)(ro + 2 * p);
            cz1.u[p] = *(const unsigned*)(rz + 32 + 2 * p);
            co1.u[p] = *(const unsigned*)(ro + 32 + 2 * p);
        }

#pragma unroll
        for (int kt = 0; kt < KT; kt += 2) {
            Frag nz0, no0, nz1, no1; // next group (kt+2, kt+3)
            if (kt + 2 < KT) {
                const int kb = (kt + 2) * 32;
#pragma unroll
                for (int p = 0; p < 8; ++p) {
                    nz0.u[p] = *(const unsigned*)(rz + kb + 2 * p);
                    no0.u[p] = *(const unsigned*)(ro + kb + 2 * p);
                    nz1.u[p] = *(const unsigned*)(rz + kb + 32 + 2 * p);
                    no1.u[p] = *(const unsigned*)(ro + kb + 32 + 2 * p);
                }
            }
            // four independent WMMA chains -> no RAW stall between issues
            accz0 = __builtin_amdgcn_wmma_f32_16x16x32_bf16(
                        false, afrag[kt].v,     false, cz0.v, (short)0, accz0,
                        false, false);
            acco0 = __builtin_amdgcn_wmma_f32_16x16x32_bf16(
                        false, afrag[kt].v,     false, co0.v, (short)0, acco0,
                        false, false);
            accz1 = __builtin_amdgcn_wmma_f32_16x16x32_bf16(
                        false, afrag[kt + 1].v, false, cz1.v, (short)0, accz1,
                        false, false);
            acco1 = __builtin_amdgcn_wmma_f32_16x16x32_bf16(
                        false, afrag[kt + 1].v, false, co1.v, (short)0, acco1,
                        false, false);
            cz0 = nz0; co0 = no0; cz1 = nz1; co1 = no1;  // rotate (no-op regs)
        }

        // ---- epilogue: bias + tanh, C/D layout: n=lm, m = r + lhi*8 -------
        const int col = n0 + lm;
        if (col < DD) {
            const float bzv = bz[col];
            const float bov = bo[col];
#pragma unroll
            for (int r = 0; r < 8; ++r) {
                size_t mrow = (size_t)(m0 + wv * 16 + lhi * 8 + r);
                zws[mrow * DD + col] = tanhf(accz0[r] + accz1[r] + bzv);
                ows[mrow * DD + col] = tanhf(acco0[r] + acco1[r] + bov);
            }
        }
        __syncthreads();   // before next N-tile overwrites ldsW
    }
}

// ---------------------------------------------------------------------------
// Kernel 1: per-chunk affine summaries.  c' = A*c + Bc over 128 steps.
// thread = (b, chunk, d); coalesced over d.
// ---------------------------------------------------------------------------
__global__ __launch_bounds__(320)
void scan_summaries_kernel(const float* __restrict__ g,
                           const float* __restrict__ zws,
                           float* __restrict__ Asum,
                           float* __restrict__ Bsum) {
    const int d = threadIdx.x;
    if (d >= DD) return;
    const int b  = blockIdx.x;
    const int ch = blockIdx.y;
    size_t base = ((size_t)b * TT + (size_t)ch * CHL) * DD + d;

    float A = 1.0f, Bc = 0.0f;
#pragma unroll 4
    for (int i = 0; i < CHL; ++i) {
        size_t off = base + (size_t)i * DD;
        float gt = g[off];
        float zt = zws[off];
        A  = gt * A;
        Bc = gt * Bc + (1.0f - gt) * zt;
    }
    int idx = (b * NCH + ch) * DD + d;
    Asum[idx] = A;
    Bsum[idx] = Bc;
}

// ---------------------------------------------------------------------------
// Kernel 2: apply.  Fold <=16 chunk prefixes inline (cheap), then replay the
// chunk emitting h = o * c.
// ---------------------------------------------------------------------------
__global__ __launch_bounds__(320)
void scan_apply_kernel(const float* __restrict__ g,
                       const float* __restrict__ zws,
                       const float* __restrict__ ows,
                       const float* __restrict__ Asum,
                       const float* __restrict__ Bsum,
                       float* __restrict__ h) {
    const int d = threadIdx.x;
    if (d >= DD) return;
    const int b  = blockIdx.x;
    const int ch = blockIdx.y;

    float c = 0.0f;
    for (int j = 0; j < ch; ++j) {
        int idx = (b * NCH + j) * DD + d;
        c = Asum[idx] * c + Bsum[idx];
    }

    size_t base = ((size_t)b * TT + (size_t)ch * CHL) * DD + d;
#pragma unroll 4
    for (int i = 0; i < CHL; ++i) {
        size_t off = base + (size_t)i * DD;
        float gt = g[off];
        float zt = zws[off];
        float ot = ows[off];
        c = gt * c + (1.0f - gt) * zt;
        h[off] = ot * c;
    }
}

// ---------------------------------------------------------------------------
extern "C" void kernel_launch(void* const* d_in, const int* in_sizes, int n_in,
                              void* d_out, int out_size, void* d_ws, size_t ws_size,
                              hipStream_t stream) {
    const float* gate = (const float*)d_in[0];   // [B,T,D]
    const float* X    = (const float*)d_in[1];   // [B,T,D]
    const float* Wz   = (const float*)d_in[2];   // [D,D]
    const float* bz   = (const float*)d_in[3];   // [D]
    const float* Wo   = (const float*)d_in[4];   // [D,D]
    const float* bo   = (const float*)d_in[5];   // [D]
    float* out = (float*)d_out;                  // [B,T,D]

    // workspace layout: z | o | Asum | Bsum
    float* zws  = (float*)d_ws;
    float* ows  = zws + (size_t)MM * DD;
    float* Asum = ows + (size_t)MM * DD;
    float* Bsum = Asum + (size_t)BB * NCH * DD;

    // GEMM: 2048 blocks x 128 threads, each owns a 64-row stripe of X.
    gated_gemm_kernel<<<MM / MROWS, 128, 0, stream>>>(X, Wz, Wo, bz, bo, zws, ows);

    // Chunked affine scan: (b, chunk) grid, d across threads.
    scan_summaries_kernel<<<dim3(BB, NCH), 320, 0, stream>>>(gate, zws, Asum, Bsum);
    scan_apply_kernel<<<dim3(BB, NCH), 320, 0, stream>>>(gate, zws, ows, Asum, Bsum, out);
}
